// Model_58179626992415
// MI455X (gfx1250) — matrix operand order repaired and checked by reading, not compiled, not run on hardware
//
#include <hip/hip_runtime.h>

typedef __attribute__((ext_vector_type(16))) __bf16 v16bf;
typedef __attribute__((ext_vector_type(8)))  __bf16 v8bf;
typedef __attribute__((ext_vector_type(8)))  float  v8f;

#define CH 128

// ---------------------------------------------------------------------------
// helpers
// ---------------------------------------------------------------------------
__device__ __forceinline__ v16bf cvt16(float4 a, float4 b, float4 c, float4 d) {
  v16bf v;
  v[0]=(__bf16)a.x;  v[1]=(__bf16)a.y;  v[2]=(__bf16)a.z;  v[3]=(__bf16)a.w;
  v[4]=(__bf16)b.x;  v[5]=(__bf16)b.y;  v[6]=(__bf16)b.z;  v[7]=(__bf16)b.w;
  v[8]=(__bf16)c.x;  v[9]=(__bf16)c.y;  v[10]=(__bf16)c.z; v[11]=(__bf16)c.w;
  v[12]=(__bf16)d.x; v[13]=(__bf16)d.y; v[14]=(__bf16)d.z; v[15]=(__bf16)d.w;
  return v;
}

// B operand: pre-swizzled contiguous 16 bf16 per (wt, kk, lane)
__device__ __forceinline__ v16bf load_b(const __bf16* wz, int KK, int wt, int kk, int lane) {
  return *(const v16bf*)(wz + ((size_t)(((wt * KK + kk) * 32) + lane) << 4));
}

// A operand from fp32 row-major global: m = lane&15, lane-half picks K chunk
__device__ __forceinline__ v16bf load_a_global(const float* row_ptr, int k0, int h) {
  const float* p = row_ptr + k0 + (h << 3);
  float4 f0 = *(const float4*)(p);
  float4 f1 = *(const float4*)(p + 4);
  float4 f2 = *(const float4*)(p + 16);
  float4 f3 = *(const float4*)(p + 20);
  return cvt16(f0, f1, f2, f3);
}

// ---------------------------------------------------------------------------
// weight conversion + swizzle into wave32 WMMA B-layout
// dst[((wt*KK+kk)*32+lane)*16 + j] = w[k*N + n],
//   n = wt*16 + (lane&15),  k = kk*32 + 8*(lane>>4) + (j&7) + 16*(j>>3)
// ---------------------------------------------------------------------------
__global__ void swizzle_w(const float* __restrict__ w, __bf16* __restrict__ dst,
                          int K, int N) {
  int tid = blockIdx.x * blockDim.x + threadIdx.x;
  if (tid >= K * N) return;
  int j    = tid & 15;
  int lane = (tid >> 4) & 31;
  int kkwt = tid >> 9;
  int KK   = K >> 5;
  int kk   = kkwt % KK;
  int wt   = kkwt / KK;
  int n = (wt << 4) + (lane & 15);
  int k = (kk << 5) + ((lane >> 4) << 3) + (j & 7) + ((j >> 3) << 4);
  dst[tid] = (__bf16)w[(size_t)k * N + n];
}

// ---------------------------------------------------------------------------
// encoder: out[row_off+r, :] = feats[r, 0:32] @ W(32x128) + bias,  K=32 -> 1 WMMA
// block = 256 threads = 8 waves; wave wt owns a 16-column tile
// ---------------------------------------------------------------------------
__global__ void encode_kernel(const float* __restrict__ feats, const __bf16* __restrict__ wz,
                              const float* __restrict__ bias, float* __restrict__ out,
                              int rows, int row_off) {
  int lane = threadIdx.x & 31;
  int wt   = threadIdx.x >> 5;
  int row0 = blockIdx.x << 4;
  int m_l = lane & 15, h = lane >> 4;
  int arow = row0 + m_l; if (arow >= rows) arow = rows - 1;
  v16bf a = load_a_global(feats + (size_t)arow * 32, 0, h);
  v16bf b = load_b(wz, 1, wt, 0, lane);
  v8f acc = {};
  acc = __builtin_amdgcn_wmma_f32_16x16x32_bf16(false, a, false, b, (short)0, acc, false, false);
  int col = (wt << 4) + m_l;
  float bb = bias[col];
  float* po = out + (size_t)(row_off + row0 + (h << 3)) * CH + col;
  if (row0 + 16 <= rows) {              // fast path: full tile, no guards
#pragma unroll
    for (int r = 0; r < 8; ++r) po[(size_t)r * CH] = acc[r] + bb;
  } else {
#pragma unroll
    for (int r = 0; r < 8; ++r) {
      int m = r + (h << 3);
      if (row0 + m < rows) po[(size_t)r * CH] = acc[r] + bb;
    }
  }
}

// ---------------------------------------------------------------------------
// x[i] = x_all[node_row_idx[i]]    (32 lanes/row, float4)
// ---------------------------------------------------------------------------
__global__ void gather_kernel(const float* __restrict__ xall, const int* __restrict__ idx,
                              float* __restrict__ out, int n) {
  int t = blockIdx.x * blockDim.x + threadIdx.x;
  int i = t >> 5;
  if (i >= n) return;
  int c = (t & 31) << 2;
  int r = idx[i];
  *(float4*)(out + (size_t)i * CH + c) = *(const float4*)(xall + (size_t)r * CH + c);
}

__global__ void zero_kernel(float* __restrict__ p, int n4) {
  int t = blockIdx.x * blockDim.x + threadIdx.x;
  if (t < n4) ((float4*)p)[t] = make_float4(0.f, 0.f, 0.f, 0.f);
}

__global__ void deg_count_kernel(const int* __restrict__ dst, float* __restrict__ deg, int nedges) {
  int e = blockIdx.x * blockDim.x + threadIdx.x;
  if (e < nedges) unsafeAtomicAdd(&deg[dst[e]], 1.0f);
}

// agg[dst] += x[src]  (32 lanes per edge, 4 channels each, hw f32 atomics)
__global__ void scatter_kernel(const float* __restrict__ x, const int* __restrict__ src,
                               const int* __restrict__ dst, float* __restrict__ agg, int nedges) {
  int t = blockIdx.x * blockDim.x + threadIdx.x;
  int e = t >> 5;
  if (e >= nedges) return;
  int c = (t & 31) << 2;
  int s = src[e], d = dst[e];
  float4 v = *(const float4*)(x + (size_t)s * CH + c);
  float* p = agg + (size_t)d * CH + c;
  unsafeAtomicAdd(p + 0, v.x);
  unsafeAtomicAdd(p + 1, v.y);
  unsafeAtomicAdd(p + 2, v.z);
  unsafeAtomicAdd(p + 3, v.w);
}

// ---------------------------------------------------------------------------
// SAGE layer: out = act(x @ Ws + (agg/deg) @ Wn + b)
// block = 256 thr = 8 waves, one 16-row tile; LDS stages bf16 A for both mats;
// the 1/max(deg,1) mean-normalization is fused into the staging stage.
// ---------------------------------------------------------------------------
__global__ void sage_gemm_kernel(const float* __restrict__ x, const float* __restrict__ agg,
                                 const float* __restrict__ deg,
                                 const __bf16* __restrict__ wzs, const __bf16* __restrict__ wzn,
                                 const float* __restrict__ bias, float* __restrict__ out,
                                 int nrows, int relu) {
  __shared__ __bf16 lx[16 * CH];
  __shared__ __bf16 lg[16 * CH];
  int tid = threadIdx.x;
  int row0 = blockIdx.x << 4;
  {
    int base = tid << 3;            // 8 elements per thread (2048 total)
    int rr = base >> 7;
    int cc = base & (CH - 1);
    int grow = row0 + rr; if (grow >= nrows) grow = nrows - 1;
    float inv = 1.0f / fmaxf(deg[grow], 1.0f);
    const float* px = x   + (size_t)grow * CH + cc;
    const float* pg = agg + (size_t)grow * CH + cc;
    float4 x0 = *(const float4*)px, x1 = *(const float4*)(px + 4);
    float4 g0 = *(const float4*)pg, g1 = *(const float4*)(pg + 4);
    v8bf vx, vg;
    vx[0]=(__bf16)x0.x; vx[1]=(__bf16)x0.y; vx[2]=(__bf16)x0.z; vx[3]=(__bf16)x0.w;
    vx[4]=(__bf16)x1.x; vx[5]=(__bf16)x1.y; vx[6]=(__bf16)x1.z; vx[7]=(__bf16)x1.w;
    vg[0]=(__bf16)(g0.x*inv); vg[1]=(__bf16)(g0.y*inv); vg[2]=(__bf16)(g0.z*inv); vg[3]=(__bf16)(g0.w*inv);
    vg[4]=(__bf16)(g1.x*inv); vg[5]=(__bf16)(g1.y*inv); vg[6]=(__bf16)(g1.z*inv); vg[7]=(__bf16)(g1.w*inv);
    *(v8bf*)(lx + base) = vx;
    *(v8bf*)(lg + base) = vg;
  }
  __syncthreads();

  int lane = tid & 31, wt = tid >> 5;
  int m_l = lane & 15, h = lane >> 4;
  v8f acc = {};
#pragma unroll
  for (int kk = 0; kk < 4; ++kk) {
    int abase = m_l * CH + (kk << 5) + (h << 3);
    v8bf xlo = *(const v8bf*)(lx + abase);
    v8bf xhi = *(const v8bf*)(lx + abase + 16);
    v16bf ax = __builtin_shufflevector(xlo, xhi, 0,1,2,3,4,5,6,7,8,9,10,11,12,13,14,15);
    v16bf bs = load_b(wzs, 4, wt, kk, lane);
    acc = __builtin_amdgcn_wmma_f32_16x16x32_bf16(false, ax, false, bs, (short)0, acc, false, false);
    v8bf glo = *(const v8bf*)(lg + abase);
    v8bf ghi = *(const v8bf*)(lg + abase + 16);
    v16bf ag = __builtin_shufflevector(glo, ghi, 0,1,2,3,4,5,6,7,8,9,10,11,12,13,14,15);
    v16bf bn = load_b(wzn, 4, wt, kk, lane);
    acc = __builtin_amdgcn_wmma_f32_16x16x32_bf16(false, ag, false, bn, (short)0, acc, false, false);
  }
  int col = (wt << 4) + m_l;
  float bb = bias[col];
  float* po = out + (size_t)(row0 + (h << 3)) * CH + col;
  if (row0 + 16 <= nrows) {             // fast path: full tile, no guards
#pragma unroll
    for (int r = 0; r < 8; ++r) {
      float v = acc[r] + bb;
      if (relu) v = fmaxf(v, 0.0f);
      po[(size_t)r * CH] = v;
    }
  } else {
#pragma unroll
    for (int r = 0; r < 8; ++r) {
      float v = acc[r] + bb;
      if (relu) v = fmaxf(v, 0.0f);
      int m = r + (h << 3);
      if (row0 + m < nrows) po[(size_t)r * CH] = v;
    }
  }
}

// ---------------------------------------------------------------------------
// head: out[50000,16] = x @ head_w(128x16) + head_b   (1 col tile per wave)
// ---------------------------------------------------------------------------
__global__ void head_kernel(const float* __restrict__ x, const __bf16* __restrict__ wz,
                            const float* __restrict__ bias, float* __restrict__ out, int nrows) {
  int lane = threadIdx.x & 31;
  int w    = threadIdx.x >> 5;
  int row0 = (blockIdx.x * 8 + w) << 4;
  if (row0 >= nrows) return;          // wave-uniform
  int m_l = lane & 15, h = lane >> 4;
  int arow = row0 + m_l; if (arow >= nrows) arow = nrows - 1;
  const float* px = x + (size_t)arow * CH;
  v8f acc = {};
#pragma unroll
  for (int kk = 0; kk < 4; ++kk) {
    v16bf a = load_a_global(px, kk << 5, h);
    v16bf b = load_b(wz, 4, 0, kk, lane);
    acc = __builtin_amdgcn_wmma_f32_16x16x32_bf16(false, a, false, b, (short)0, acc, false, false);
  }
  float bb = bias[m_l];
  float* po = out + (size_t)(row0 + (h << 3)) * 16 + m_l;
  if (row0 + 16 <= nrows) {             // fast path
#pragma unroll
    for (int r = 0; r < 8; ++r) po[r * 16] = acc[r] + bb;
  } else {
#pragma unroll
    for (int r = 0; r < 8; ++r) {
      int m = r + (h << 3);
      if (row0 + m < nrows) po[r * 16] = acc[r] + bb;
    }
  }
}

// ---------------------------------------------------------------------------
// launcher
// ---------------------------------------------------------------------------
extern "C" void kernel_launch(void* const* d_in, const int* in_sizes, int n_in,
                              void* d_out, int out_size, void* d_ws, size_t ws_size,
                              hipStream_t stream) {
  const float* feats0   = (const float*)d_in[0];
  const float* feats1   = (const float*)d_in[1];
  const float* enc_w0   = (const float*)d_in[2];
  const float* enc_b0   = (const float*)d_in[3];
  const float* enc_w1   = (const float*)d_in[4];
  const float* enc_b1   = (const float*)d_in[5];
  const float* w_self0  = (const float*)d_in[6];
  const float* w_neigh0 = (const float*)d_in[7];
  const float* b0       = (const float*)d_in[8];
  const float* w_self1  = (const float*)d_in[9];
  const float* w_neigh1 = (const float*)d_in[10];
  const float* b1       = (const float*)d_in[11];
  const float* head_w   = (const float*)d_in[12];
  const float* head_b   = (const float*)d_in[13];
  const int*   nri      = (const int*)d_in[14];
  const int*   ei       = (const int*)d_in[15];

  const int nodes = in_sizes[14];
  const int edges = in_sizes[15] / 2;
  const int t0    = in_sizes[0] / 32;
  const int t1    = in_sizes[1] / 32;
  const int* src  = ei;
  const int* dstp = ei + edges;

  char* ws = (char*)d_ws;
  size_t off = 0;
  auto alloc = [&](size_t bytes) -> void* {
    void* p = ws + off;
    off += (bytes + 255) & ~(size_t)255;
    return p;
  };
  float*  xbuf   = (float*)alloc((size_t)nodes * CH * sizeof(float));
  float*  aggbuf = (float*)alloc((size_t)nodes * CH * sizeof(float));
  float*  hbuf   = (float*)alloc((size_t)nodes * CH * sizeof(float));
  float*  deg    = (float*)alloc((size_t)nodes * sizeof(float));
  __bf16* wz_s0  = (__bf16*)alloc((size_t)CH * CH * sizeof(__bf16));
  __bf16* wz_n0  = (__bf16*)alloc((size_t)CH * CH * sizeof(__bf16));
  __bf16* wz_s1  = (__bf16*)alloc((size_t)CH * CH * sizeof(__bf16));
  __bf16* wz_n1  = (__bf16*)alloc((size_t)CH * CH * sizeof(__bf16));
  __bf16* wz_hd  = (__bf16*)alloc((size_t)CH * 16 * sizeof(__bf16));
  __bf16* wz_e0  = (__bf16*)alloc((size_t)32 * CH * sizeof(__bf16));
  __bf16* wz_e1  = (__bf16*)alloc((size_t)32 * CH * sizeof(__bf16));

  const int TB = 256;
  auto sw = [&](const float* w, __bf16* dw, int K, int N) {
    int total = K * N;
    swizzle_w<<<(total + TB - 1) / TB, TB, 0, stream>>>(w, dw, K, N);
  };
  sw(w_self0, wz_s0, CH, CH);
  sw(w_neigh0, wz_n0, CH, CH);
  sw(w_self1, wz_s1, CH, CH);
  sw(w_neigh1, wz_n1, CH, CH);
  sw(head_w, wz_hd, CH, 16);
  sw(enc_w0, wz_e0, 32, CH);
  sw(enc_w1, wz_e1, 32, CH);

  // encode both tables into aggbuf (used as x_all scratch)
  encode_kernel<<<(t0 + 15) / 16, TB, 0, stream>>>(feats0, wz_e0, enc_b0, aggbuf, t0, 0);
  encode_kernel<<<(t1 + 15) / 16, TB, 0, stream>>>(feats1, wz_e1, enc_b1, aggbuf, t1, t0);

  // gather x = x_all[node_row_idx]
  gather_kernel<<<((nodes * 32) + TB - 1) / TB, TB, 0, stream>>>(aggbuf, nri, xbuf, nodes);

  // degree
  zero_kernel<<<((nodes / 4) + TB - 1) / TB, TB, 0, stream>>>(deg, nodes / 4);
  deg_count_kernel<<<(edges + TB - 1) / TB, TB, 0, stream>>>(dstp, deg, edges);

  const int n4   = nodes * CH / 4;
  const int sthr = edges * 32;

  // layer 0
  zero_kernel<<<(n4 + TB - 1) / TB, TB, 0, stream>>>(aggbuf, n4);
  scatter_kernel<<<(sthr + TB - 1) / TB, TB, 0, stream>>>(xbuf, src, dstp, aggbuf, edges);
  sage_gemm_kernel<<<(nodes + 15) / 16, TB, 0, stream>>>(xbuf, aggbuf, deg, wz_s0, wz_n0, b0, hbuf, nodes, 1);

  // layer 1
  zero_kernel<<<(n4 + TB - 1) / TB, TB, 0, stream>>>(aggbuf, n4);
  scatter_kernel<<<(sthr + TB - 1) / TB, TB, 0, stream>>>(hbuf, src, dstp, aggbuf, edges);
  sage_gemm_kernel<<<(nodes + 15) / 16, TB, 0, stream>>>(hbuf, aggbuf, deg, wz_s1, wz_n1, b1, xbuf, nodes, 0);

  // head
  head_kernel<<<(nodes + 127) / 128, TB, 0, stream>>>(xbuf, wz_hd, head_b, (float*)d_out, nodes);
}
